// Model_50465865728485
// MI455X (gfx1250) — matrix operand order repaired and checked by reading, not compiled, not run on hardware
//
#include <hip/hip_runtime.h>
#include <hip/hip_bf16.h>
#include <math.h>

typedef __attribute__((ext_vector_type(16))) _Float16 v16h;
typedef __attribute__((ext_vector_type(8)))  _Float16 v8h;
typedef __attribute__((ext_vector_type(8)))  float    v8f;

#define B_   8
#define CIN  7
#define SEQ  336
#define PRED 96
#define LTOT 432
#define DM   256
#define NH   8
#define DH   32
#define DFF  1024
#define NSEQ (B_*CIN)        /* 56 */
#define TOK  (NSEQ*LTOT)     /* 24192 */
#define EPS_ 1e-5f

// ---------------- wave32 helpers ----------------
__device__ __forceinline__ float wsum(float x){
#pragma unroll
  for (int o = 16; o >= 1; o >>= 1) x += __shfl_xor(x, o, 32);
  return x;
}
__device__ __forceinline__ float wmaxf(float x){
#pragma unroll
  for (int o = 16; o >= 1; o >>= 1) x = fmaxf(x, __shfl_xor(x, o, 32));
  return x;
}

// A-fragment (16x32 f16): lane's halves are two contiguous 16B runs at +hi*16 and +hi*16+8
__device__ __forceinline__ v16h load_afrag(const _Float16* __restrict__ rowp, int hi){
  v8h a0 = *(const v8h*)(rowp + hi*16);
  v8h a1 = *(const v8h*)(rowp + hi*16 + 8);
  v16h a;
#pragma unroll
  for (int i = 0; i < 8; ++i) { a[i] = a0[i]; a[8+i] = a1[i]; }
  return a;
}

// ---------------- front-end kernels ----------------
__global__ void k_stats(const float* __restrict__ x_enc, float* __restrict__ means,
                        float* __restrict__ stdev)
{
  int idx = threadIdx.x;
  if (idx >= NSEQ) return;
  int b = idx / CIN, c = idx % CIN;
  float s = 0.f, q = 0.f;
  for (int t = 0; t < SEQ; ++t) {
    float v = x_enc[((size_t)b*SEQ + t)*CIN + c];
    s += v; q += v*v;
  }
  float m = s / (float)SEQ;
  float var = q / (float)SEQ - m*m;
  means[idx] = m;
  stdev[idx] = sqrtf(var + EPS_);
}

__global__ void k_xdn(const float* __restrict__ x_enc, const float* __restrict__ x_dec,
                      const float* __restrict__ means, const float* __restrict__ stdev,
                      float* __restrict__ xdn)
{
  int tid = blockIdx.x*blockDim.x + threadIdx.x;
  if (tid >= B_*LTOT*CIN) return;
  int c = tid % CIN; int rem = tid / CIN;
  int t = rem % LTOT; int b = rem / LTOT;
  float v;
  if (t < SEQ) v = (x_enc[((size_t)b*SEQ + t)*CIN + c] - means[b*CIN+c]) / stdev[b*CIN+c];
  else         v = x_dec[((size_t)b*LTOT + t)*CIN + c];   // zeros by construction
  xdn[tid] = v;
}

__global__ void k_bound(const float* __restrict__ xdn, const float* __restrict__ wq,
                        const float* __restrict__ wk, float* __restrict__ p)
{
  int tid = blockIdx.x*blockDim.x + threadIdx.x;
  if (tid >= B_*LTOT) return;
  int b = tid / LTOT, t = tid % LTOT;
  if (t == 0) { p[tid] = 1.0f; return; }
  const float* xq = xdn + ((size_t)b*LTOT + (t-1))*CIN;
  const float* xk = xdn + ((size_t)b*LTOT + t)*CIN;
  float dot = 0.f, nq = 0.f, nk = 0.f;
#pragma unroll
  for (int j = 0; j < CIN; ++j) {
    float qv = 0.f, kv = 0.f;
#pragma unroll
    for (int i = 0; i < CIN; ++i) { qv += xq[i]*wq[i*CIN+j]; kv += xk[i]*wk[i*CIN+j]; }
    dot += qv*kv; nq += qv*qv; nk += kv*kv;
  }
  float cs = dot / (sqrtf(nq)*sqrtf(nk) + 1e-6f);
  float pv = (1.0f - cs) * 0.5f;
  p[tid] = fminf(fmaxf(pv, 0.0f), 1.0f);
}

__global__ void k_scan(const float* __restrict__ p, float* __restrict__ pe,
                       int* __restrict__ dest, int* __restrict__ cidx)
{
  int b = threadIdx.x;
  if (b >= B_) return;
  int cs = 0;
  for (int t = 0; t < LTOT; ++t) {
    float pv = p[b*LTOT+t];
    bool m = (t == 0) || (pv >= 0.5f && t < SEQ);
    if (m) ++cs;
    dest[b*LTOT+t] = m ? (cs-1) : -1;
    int ci = cs - 1; if (ci < 0) ci = 0; if (ci > LTOT-1) ci = LTOT-1;
    cidx[b*LTOT+t] = ci;
    pe[b*LTOT+t] = fminf(fmaxf(pv, 1e-4f), 1.0f - 1e-4f);
  }
}

__global__ void k_zero(float* __restrict__ X, int n)
{
  int tid = blockIdx.x*blockDim.x + threadIdx.x;
  if (tid < n) X[tid] = 0.f;
}

__global__ void k_scatter(const float* __restrict__ xdn, const int* __restrict__ dest,
                          const float* __restrict__ w_emb, const float* __restrict__ b_emb,
                          float* __restrict__ X)
{
  int bt = blockIdx.x;                 // b*LTOT + t
  int b = bt / LTOT;
  int j = dest[bt];
  if (j < 0) return;
  int d = threadIdx.x;
  float we = w_emb[d], be = b_emb[d];
#pragma unroll
  for (int c = 0; c < CIN; ++c) {
    float xv = xdn[(size_t)bt*CIN + c];
    X[(((size_t)(b*CIN + c))*LTOT + j)*DM + d] = xv*we + be;
  }
}

// Convert weights to f16 in WMMA B-fragment-swizzled order:
// dst[ l*K*N + ((tn*(K/32)+kc)*32 + lane)*16 + e ] = W[k = kc*32+(lane>>4)*16+e][n = tn*16+(lane&15)]
// srcT=0: src[k*N + n] ; srcT=1: src[n*K + k]
__global__ void k_cvt_sw(const float* __restrict__ src, _Float16* __restrict__ dst,
                         int K, int N, int layers, int srcT)
{
  int tid = blockIdx.x*blockDim.x + threadIdx.x;
  int per = K*N;
  if (tid >= layers*per) return;
  int l = tid / per; int idx = tid % per;
  int e = idx & 15;
  int lane = (idx >> 4) & 31;
  int rest = idx >> 9;
  int kc = rest % (K >> 5);
  int tn = rest / (K >> 5);
  int k = kc*32 + (lane >> 4)*16 + e;
  int n = tn*16 + (lane & 15);
  float v = srcT ? src[(size_t)l*per + (size_t)n*K + k]
                 : src[(size_t)l*per + (size_t)k*N + n];
  dst[tid] = (_Float16)v;
}

__global__ void k_f16(const float* __restrict__ src, _Float16* __restrict__ dst, int n)
{
  int tid = blockIdx.x*blockDim.x + threadIdx.x;
  if (tid < n) dst[tid] = (_Float16)src[tid];
}

// ---------------- WMMA GEMM: C[M,N] = A[M,K](f16 row) @ Bsw[K,N](f16 swizzled) + bias ----
// MODE 0: fp32 out; MODE 1: exact-gelu -> f16 out; MODE 2: f16 out
template<int MODE>
__global__ void k_gemm(const _Float16* __restrict__ A, const _Float16* __restrict__ Bw,
                       const float* __restrict__ bias, void* __restrict__ outp,
                       int M, int N, int K)
{
  int wid = blockIdx.x*(blockDim.x >> 5) + (threadIdx.x >> 5);
  int tnc = N >> 4;
  int tiles = (M >> 4) * tnc;
  if (wid >= tiles) return;
  int tm = wid / tnc, tn = wid % tnc;
  int lane = threadIdx.x & 31;
  int lo = lane & 15, hi = lane >> 4;
  int kchunks = K >> 5;
  v8f acc = {};
  const _Float16* arow  = A  + (size_t)(tm*16 + lo)*K;
  const _Float16* bbase = Bw + ((size_t)tn*kchunks)*512 + lane*16;
  for (int kc = 0; kc < kchunks; ++kc) {
    v16h a = load_afrag(arow + kc*32, hi);
    v16h b = *(const v16h*)(bbase + (size_t)kc*512);
    acc = __builtin_amdgcn_wmma_f32_16x16x32_f16(false, a, false, b, (short)0, acc, false, false);
  }
  int nc = tn*16 + lo;
  float bb = bias[nc];
#pragma unroll
  for (int v = 0; v < 8; ++v) {
    size_t off = (size_t)(tm*16 + v + 8*hi)*N + nc; // C-frag: row v+8*hi, col lo
    float val = acc[v] + bb;
    if (MODE == 0) {
      ((float*)outp)[off] = val;
    } else if (MODE == 1) {
      float gx = 0.5f*val*(1.0f + erff(val*0.70710678118654752f));
      ((_Float16*)outp)[off] = (_Float16)gx;
    } else {
      ((_Float16*)outp)[off] = (_Float16)val;
    }
  }
}

// ---------------- fused attention: one wave per (seq, head, 16-query block) ----------------
__global__ void __launch_bounds__(32)
k_attn(const _Float16* __restrict__ Q, const _Float16* __restrict__ Kk,
       const _Float16* __restrict__ V, _Float16* __restrict__ O)
{
  __shared__ float sp[16*448];                      // 28 KB prob tile (432 keys + 16 pad)
  int id = blockIdx.x;
  int qb = id % 27; id /= 27;
  int h  = id % NH; int seq = id / NH;
  int lane = threadIdx.x;
  int lo = lane & 15, hi = lane >> 4;
  size_t base = (size_t)seq*LTOT*DM + h*DH;

  // Q fragment (16x32, K-dim = dh): contiguous vector loads
  v16h aq = load_afrag(Q + base + (size_t)(qb*16 + lo)*DM, hi);

  // scores: S[16q,16k] = Q(16x32) @ K^T(32x16) per key tile
  for (int kt = 0; kt < 27; ++kt) {
    // B-frag: B[kk][n] = K[key n][kk] -> lane reads 16 contiguous halves at +hi*16
    const _Float16* kp = Kk + base + (size_t)(kt*16 + lo)*DM;
    v8h b0 = *(const v8h*)(kp + hi*16);
    v8h b1 = *(const v8h*)(kp + hi*16 + 8);
    v16h bk;
#pragma unroll
    for (int i = 0; i < 8; ++i) { bk[i] = b0[i]; bk[8+i] = b1[i]; }
    v8f s = {};
    s = __builtin_amdgcn_wmma_f32_16x16x32_f16(false, aq, false, bk, (short)0, s, false, false);
#pragma unroll
    for (int v = 0; v < 8; ++v)
      sp[(v + 8*hi)*448 + kt*16 + lo] = s[v]*0.17677669529663687f; // 1/sqrt(32)
  }
  __syncthreads();
  // row softmax (432 valid keys), zero the pad columns
  for (int r = 0; r < 16; ++r) {
    float* row = sp + r*448;
    float mx = -1e30f;
    for (int c = lane; c < LTOT; c += 32) mx = fmaxf(mx, row[c]);
    mx = wmaxf(mx);
    float sm = 0.f;
    for (int c = lane; c < LTOT; c += 32) { float e = __expf(row[c]-mx); row[c] = e; sm += e; }
    sm = wsum(sm);
    float inv = 1.0f/sm;
    for (int c = lane; c < LTOT; c += 32) row[c] *= inv;
    if (lane < 16) row[LTOT + lane] = 0.f;
  }
  __syncthreads();
  // O(16x32) = P(16x448) @ V(448x32), two 16-wide N tiles
#pragma unroll
  for (int nb = 0; nb < 2; ++nb) {
    v8f acc = {};
    for (int kc = 0; kc < 448; kc += 32) {
      v16h a, bv;
#pragma unroll
      for (int v = 0; v < 8; ++v) {
        int ka = kc + hi*16 + 2*v;
        a[2*v]   = (_Float16)sp[lo*448 + ka];
        a[2*v+1] = (_Float16)sp[lo*448 + ka + 1];
        int kb = kc + hi*16 + 2*v;
        bv[2*v]   = (kb   < LTOT) ? V[base + (size_t)kb*DM     + nb*16 + lo] : (_Float16)0.f;
        bv[2*v+1] = (kb+1 < LTOT) ? V[base + (size_t)(kb+1)*DM + nb*16 + lo] : (_Float16)0.f;
      }
      acc = __builtin_amdgcn_wmma_f32_16x16x32_f16(false, a, false, bv, (short)0, acc, false, false);
    }
#pragma unroll
    for (int v = 0; v < 8; ++v)
      O[base + (size_t)(qb*16 + v + 8*hi)*DM + nb*16 + lo] = (_Float16)acc[v];
  }
}

// ---------------- residual + layernorm, writes fp32 X and f16 Xh ----------------
__global__ void k_resln(float* __restrict__ X, const float* __restrict__ S,
                        const float* __restrict__ g, const float* __restrict__ be,
                        _Float16* __restrict__ Xh)
{
  __shared__ float rs[8], rq[8];
  int row = blockIdx.x;
  int tid = threadIdx.x;
  size_t off = (size_t)row*DM + tid;
  float x = X[off] + S[off];
  float s = wsum(x), q = wsum(x*x);
  if ((tid & 31) == 0) { rs[tid>>5] = s; rq[tid>>5] = q; }
  __syncthreads();
  float ts = 0.f, tq = 0.f;
#pragma unroll
  for (int i = 0; i < 8; ++i) { ts += rs[i]; tq += rq[i]; }
  float m   = ts * (1.0f/DM);
  float var = tq * (1.0f/DM) - m*m;
  float y = (x - m) * rsqrtf(var + EPS_) * g[tid] + be[tid];
  X[off]  = y;
  Xh[off] = (_Float16)y;
}

// ---------------- batchnorm stats over all 24192 rows ----------------
__global__ void k_bnstats(const float* __restrict__ X, const float* __restrict__ g,
                          float* __restrict__ mu, float* __restrict__ sg)
{
  __shared__ float rs[8], rq[8];
  int d = blockIdx.x;
  float s = 0.f, q = 0.f;
  for (int r = threadIdx.x; r < TOK; r += 256) {
    float v = X[(size_t)r*DM + d];
    s += v; q += v*v;
  }
  float a = wsum(s), b = wsum(q);
  if ((threadIdx.x & 31) == 0) { rs[threadIdx.x>>5] = a; rq[threadIdx.x>>5] = b; }
  __syncthreads();
  if (threadIdx.x == 0) {
    float ts = 0.f, tq = 0.f;
    for (int i = 0; i < 8; ++i) { ts += rs[i]; tq += rq[i]; }
    float m = ts / (float)TOK;
    float var = tq / (float)TOK - m*m;
    mu[d] = m;
    sg[d] = g[d] * rsqrtf(var + EPS_);
  }
}

// group means of normalized enc_out: gbar[b,t,c] = mean_d BN(X[b*774144 + t*1792 + c*256 + d])
__global__ void k_gbar(const float* __restrict__ X, const float* __restrict__ mu,
                       const float* __restrict__ sg, const float* __restrict__ bb,
                       float* __restrict__ gbar)
{
  int wid = blockIdx.x*8 + (threadIdx.x >> 5);
  if (wid >= B_*LTOT*CIN) return;
  int lane = threadIdx.x & 31;
  int c = wid % CIN; int rem = wid / CIN;
  int t = rem % LTOT; int b = rem / LTOT;
  size_t offx = (size_t)b*(CIN*LTOT*DM) + (size_t)t*(CIN*DM) + (size_t)c*DM;
  float acc = 0.f;
  for (int d = lane; d < DM; d += 32)
    acc += sg[d]*(X[offx+d] - mu[d]) + bb[d];
  acc = wsum(acc);
  if (lane == 0) gbar[wid] = acc * (1.0f/DM);
}

// EMA associative scan per (b,c), de-normalize, emit last PRED steps
__global__ void k_decode(const float* __restrict__ gbar, const float* __restrict__ pe,
                         const int* __restrict__ cidx, const float* __restrict__ stdev,
                         const float* __restrict__ means, float* __restrict__ out)
{
  int idx = threadIdx.x;
  if (idx >= NSEQ) return;
  int b = idx / CIN, c = idx % CIN;
  float h = 0.f;
  for (int t = 0; t < LTOT; ++t) {
    float z  = gbar[((size_t)b*LTOT + cidx[b*LTOT+t])*CIN + c];
    float pp = pe[b*LTOT+t];
    h = (1.0f - pp)*h + pp*z;
    if (t >= SEQ) out[((size_t)b*PRED + (t-SEQ))*CIN + c] = h*stdev[idx] + means[idx];
  }
}

// ---------------- host orchestration ----------------
extern "C" void kernel_launch(void* const* d_in, const int* in_sizes, int n_in,
                              void* d_out, int out_size, void* d_ws, size_t ws_size,
                              hipStream_t stream)
{
  (void)in_sizes; (void)n_in; (void)out_size; (void)ws_size;
  const float* x_enc = (const float*)d_in[0];
  const float* x_dec = (const float*)d_in[2];
  const float* wq_r  = (const float*)d_in[4];
  const float* wk_r  = (const float*)d_in[5];
  const float* w_emb = (const float*)d_in[6];
  const float* b_emb = (const float*)d_in[7];
  const float* Wq    = (const float*)d_in[8];
  const float* bq    = (const float*)d_in[9];
  const float* Wk    = (const float*)d_in[10];
  const float* bk    = (const float*)d_in[11];
  const float* Wv    = (const float*)d_in[12];
  const float* bv    = (const float*)d_in[13];
  const float* Wo    = (const float*)d_in[14];
  const float* bo    = (const float*)d_in[15];
  const float* w1    = (const float*)d_in[16];
  const float* b1    = (const float*)d_in[17];
  const float* w2    = (const float*)d_in[18];
  const float* b2    = (const float*)d_in[19];
  const float* g1    = (const float*)d_in[20];
  const float* be1   = (const float*)d_in[21];
  const float* g2    = (const float*)d_in[22];
  const float* be2   = (const float*)d_in[23];
  const float* bn_g  = (const float*)d_in[24];
  const float* bn_b  = (const float*)d_in[25];
  float* out = (float*)d_out;

  // workspace carve-out (all offsets 256B aligned)
  char* wsb = (char*)d_ws;
  size_t off = 0;
  auto alloc = [&](size_t bytes) -> void* {
    void* p = wsb + off;
    off += (bytes + 255) & ~(size_t)255;
    return p;
  };
  const size_t TD = (size_t)TOK*DM;            // 6,193,152
  float*    X   = (float*)   alloc(TD*4);
  float*    S   = (float*)   alloc(TD*4);
  _Float16* Qh  = (_Float16*)alloc(TD*2);
  _Float16* Kh  = (_Float16*)alloc(TD*2);
  _Float16* Vh  = (_Float16*)alloc(TD*2);
  _Float16* Xh  = (_Float16*)alloc(TD*2);
  _Float16* Yh  = (_Float16*)alloc((size_t)TOK*DFF*2);
  _Float16* WqH = (_Float16*)alloc((size_t)2*DM*DM*2);
  _Float16* WkH = (_Float16*)alloc((size_t)2*DM*DM*2);
  _Float16* WvH = (_Float16*)alloc((size_t)2*DM*DM*2);
  _Float16* WoH = (_Float16*)alloc((size_t)2*DM*DM*2);
  _Float16* W1H = (_Float16*)alloc((size_t)2*DFF*DM*2);
  _Float16* W2H = (_Float16*)alloc((size_t)2*DM*DFF*2);
  float*    xdn   = (float*)alloc((size_t)B_*LTOT*CIN*4);
  float*    parr  = (float*)alloc((size_t)B_*LTOT*4);
  float*    pe    = (float*)alloc((size_t)B_*LTOT*4);
  int*      dest  = (int*)  alloc((size_t)B_*LTOT*4);
  int*      cidx  = (int*)  alloc((size_t)B_*LTOT*4);
  float*    means = (float*)alloc(NSEQ*4);
  float*    stdev = (float*)alloc(NSEQ*4);
  float*    mu    = (float*)alloc(DM*4);
  float*    sg    = (float*)alloc(DM*4);
  float*    gbar  = (float*)alloc((size_t)B_*LTOT*CIN*4);

  // ---- front end ----
  k_stats<<<1, 64, 0, stream>>>(x_enc, means, stdev);
  k_xdn<<<(B_*LTOT*CIN + 255)/256, 256, 0, stream>>>(x_enc, x_dec, means, stdev, xdn);
  k_bound<<<(B_*LTOT + 127)/128, 128, 0, stream>>>(xdn, wq_r, wk_r, parr);
  k_scan<<<1, 8, 0, stream>>>(parr, pe, dest, cidx);
  k_zero<<<(int)((TD + 255)/256), 256, 0, stream>>>(X, (int)TD);
  k_scatter<<<B_*LTOT, 256, 0, stream>>>(xdn, dest, w_emb, b_emb, X);

  // ---- weights -> f16, pre-swizzled to WMMA B-fragment order ----
  k_cvt_sw<<<(2*DM*DM + 255)/256, 256, 0, stream>>>(Wq, WqH, DM, DM, 2, 0);
  k_cvt_sw<<<(2*DM*DM + 255)/256, 256, 0, stream>>>(Wk, WkH, DM, DM, 2, 0);
  k_cvt_sw<<<(2*DM*DM + 255)/256, 256, 0, stream>>>(Wv, WvH, DM, DM, 2, 0);
  k_cvt_sw<<<(2*DM*DM + 255)/256, 256, 0, stream>>>(Wo, WoH, DM, DM, 2, 0);
  k_cvt_sw<<<(2*DFF*DM + 255)/256, 256, 0, stream>>>(w1, W1H, DM, DFF, 2, 1); // y=x@w1^T
  k_cvt_sw<<<(2*DM*DFF + 255)/256, 256, 0, stream>>>(w2, W2H, DFF, DM, 2, 1); // o=y@w2^T

  k_f16<<<(int)((TD + 255)/256), 256, 0, stream>>>(X, Xh, (int)TD);

  auto gemm_blocks = [](int M, int N){ return ((M/16)*(N/16) + 7)/8; };

  // ---- encoder layers ----
  for (int l = 0; l < 2; ++l) {
    k_gemm<2><<<gemm_blocks(TOK,DM), 256, 0, stream>>>(Xh, WqH + (size_t)l*DM*DM, bq + l*DM, Qh, TOK, DM, DM);
    k_gemm<2><<<gemm_blocks(TOK,DM), 256, 0, stream>>>(Xh, WkH + (size_t)l*DM*DM, bk + l*DM, Kh, TOK, DM, DM);
    k_gemm<2><<<gemm_blocks(TOK,DM), 256, 0, stream>>>(Xh, WvH + (size_t)l*DM*DM, bv + l*DM, Vh, TOK, DM, DM);
    k_attn<<<NSEQ*NH*27, 32, 0, stream>>>(Qh, Kh, Vh, Xh);          // attention out -> Xh (f16)
    k_gemm<0><<<gemm_blocks(TOK,DM), 256, 0, stream>>>(Xh, WoH + (size_t)l*DM*DM, bo + l*DM, S, TOK, DM, DM);
    k_resln<<<TOK, 256, 0, stream>>>(X, S, g1 + l*DM, be1 + l*DM, Xh);
    k_gemm<1><<<gemm_blocks(TOK,DFF), 256, 0, stream>>>(Xh, W1H + (size_t)l*DFF*DM, b1 + l*DFF, Yh, TOK, DFF, DM);
    k_gemm<0><<<gemm_blocks(TOK,DM), 256, 0, stream>>>(Yh, W2H + (size_t)l*DM*DFF, b2 + l*DM, S, TOK, DM, DFF);
    k_resln<<<TOK, 256, 0, stream>>>(X, S, g2 + l*DM, be2 + l*DM, Xh);
  }

  // ---- batchnorm stats + decode ----
  k_bnstats<<<DM, 256, 0, stream>>>(X, bn_g, mu, sg);
  k_gbar<<<(B_*LTOT*CIN + 7)/8, 256, 0, stream>>>(X, mu, sg, bn_b, gbar);
  k_decode<<<1, 64, 0, stream>>>(gbar, pe, cidx, stdev, means, out);
}
